// MemoryBlock_12979391168580
// MI455X (gfx1250) — compile-verified, matrix operand order checked by reading, not run on hardware
//
#include <hip/hip_runtime.h>
#include <hip/hip_bf16.h>

// ------------------------------------------------------------------
// Problem dims (fixed by the reference)
// ------------------------------------------------------------------
#define Bsz 4
#define Ssz 1024
#define Hsz 1024
#define Msz 32768
#define ROWS (Bsz * Ssz)          // 4096 query rows
#define QSCALE 0.03125f           // 1/sqrt(1024)

// ------------------------------------------------------------------
// bf16 WMMA plumbing (gfx1250: v_wmma_f32_16x16x32_bf16, wave32)
// ------------------------------------------------------------------
typedef __bf16 bf16_t;
typedef bf16_t v16bf __attribute__((ext_vector_type(16)));
typedef bf16_t v8bf  __attribute__((ext_vector_type(8)));
typedef float  v8f   __attribute__((ext_vector_type(8)));
typedef unsigned int u32x4 __attribute__((ext_vector_type(4)));
typedef int          i32x8 __attribute__((ext_vector_type(8)));
typedef int          i32x4 __attribute__((ext_vector_type(4)));

__device__ __forceinline__ unsigned short f2bf(float f) {
  unsigned u = __float_as_uint(f);
  u += 0x7fffu + ((u >> 16) & 1u);        // round-to-nearest-even
  return (unsigned short)(u >> 16);
}

// 16-bit A/B fragment (16x32 / 32x16): per ISA, lane half 0 holds K{0..7,16..23},
// half 1 holds K{8..15,24..31} => two contiguous 16B runs per lane.
__device__ __forceinline__ v16bf load_frag(const unsigned short* p, int half) {
  v8bf lo = *reinterpret_cast<const v8bf*>(p + half * 8);
  v8bf hi = *reinterpret_cast<const v8bf*>(p + 16 + half * 8);
  return __builtin_shufflevector(lo, hi, 0,1,2,3,4,5,6,7,8,9,10,11,12,13,14,15);
}

__device__ __forceinline__ v8f wmma_bf16(v16bf a, v16bf b, v8f c) {
  return __builtin_amdgcn_wmma_f32_16x16x32_bf16(false, a, false, b, (short)0, c,
                                                 false, false);
}

// ------------------------------------------------------------------
// Tensor Data Mover: 1-D contiguous tile (nelem bf16 elements) -> LDS.
// D# packed per CDNA5 ISA §8 (group0: count/lds_addr/global_addr/type=2;
// group1: data_size=2B, tensor_dim0=nelem, tensor_dim1=1, tile_dim0=nelem,
// tile_dim1=1, dim0_stride=nelem). Trailing groups zero (<=2-D tensor).
// 6-arg builtin (clang-23 / therock): (u32x4, i32x8, i32x4, i32x4, i32x8, cpol).
// Issue from ONE wave; TENSORcnt waited in that wave, then block barrier.
// ------------------------------------------------------------------
__device__ __forceinline__ void tdm_load_1d(unsigned lds_off,
                                            const unsigned short* gsrc,
                                            unsigned nelem) {
  unsigned long long ga = (unsigned long long)(size_t)gsrc;
  u32x4 g0;
  g0[0] = 1u;                                        // count=1 (user descriptor)
  g0[1] = lds_off;                                   // lds_addr (bytes)
  g0[2] = (unsigned)(ga & 0xFFFFFFFFu);              // global_addr[31:0]
  g0[3] = (unsigned)((ga >> 32) & 0x01FFFFFFu) | (2u << 30);  // ga[56:32] | type=2
  i32x8 g1;
  g1[0] = (int)(1u << 16);                           // data_size=1 => 2 bytes
  g1[1] = (int)((nelem & 0xFFFFu) << 16);            // tensor_dim0[15:0] @bit48
  g1[2] = (int)(((nelem >> 16) & 0xFFFFu) | (1u << 16)); // dim0[31:16] | dim1=1
  g1[3] = (int)((nelem & 0xFFFFu) << 16);            // tile_dim0 @bit112
  g1[4] = 1;                                         // tile_dim1=1, tile_dim2=0
  g1[5] = (int)nelem;                                // tensor_dim0_stride[31:0]
  g1[6] = 0;                                         // stride hi | dim1_stride lo
  g1[7] = 0;
  i32x4 z4 = {0, 0, 0, 0};
  i32x8 z8 = {0, 0, 0, 0, 0, 0, 0, 0};
  __builtin_amdgcn_tensor_load_to_lds(g0, g1, z4, z4, z8, 0);
}

// ------------------------------------------------------------------
// Elementwise converts
// ------------------------------------------------------------------
__global__ void cvt_bf16_kernel(const float* __restrict__ in,
                                unsigned short* __restrict__ out, size_t n) {
  for (size_t i = (size_t)blockIdx.x * blockDim.x + threadIdx.x; i < n;
       i += (size_t)gridDim.x * blockDim.x)
    out[i] = f2bf(in[i]);
}

// memory_values [M][H] f32  ->  Vt [H][M] bf16 (tiled transpose)
__global__ void transpose_cvt_kernel(const float* __restrict__ V,
                                     unsigned short* __restrict__ Vt) {
  __shared__ float t[32][33];
  int bx = blockIdx.x;            // along M (M/32 blocks)
  int by = blockIdx.y;            // along H (H/32 blocks)
  int x = threadIdx.x, y = threadIdx.y;       // 32 x 8
  for (int yy = y; yy < 32; yy += 8)
    t[yy][x] = V[(size_t)(bx * 32 + yy) * Hsz + by * 32 + x];
  __syncthreads();
  for (int yy = y; yy < 32; yy += 8)
    Vt[(size_t)(by * 32 + yy) * Msz + bx * 32 + x] = f2bf(t[x][yy]);
}

// ------------------------------------------------------------------
// GEMM: C[r][j] = sum_d A[r][d]*Bw[j][d] + bias[j]   (both K-contiguous)
// grid.x = rows/16, block = 256 (8 waves); wave w -> cols [w*128, w*128+128)
// ------------------------------------------------------------------
template <int OUT_BF16>
__global__ __launch_bounds__(256) void gemm_kernel(
    const unsigned short* __restrict__ A, const unsigned short* __restrict__ Bw,
    const float* __restrict__ bias, unsigned short* __restrict__ outb,
    float* __restrict__ outf) {
  const int tid = threadIdx.x, wave = tid >> 5, lane = tid & 31;
  const int half = lane >> 4, l16 = lane & 15;
  const int row0 = blockIdx.x * 16;
  const int c0 = wave * 128;
  v8f acc[8];
  #pragma unroll
  for (int n = 0; n < 8; ++n) acc[n] = (v8f){0,0,0,0,0,0,0,0};
  const unsigned short* arow = A + (size_t)(row0 + l16) * Hsz;
  for (int k0 = 0; k0 < Hsz; k0 += 32) {
    v16bf a = load_frag(arow + k0, half);
    #pragma unroll
    for (int n = 0; n < 8; ++n) {
      v16bf b = load_frag(Bw + (size_t)(c0 + n * 16 + l16) * Hsz + k0, half);
      acc[n] = wmma_bf16(a, b, acc[n]);
    }
  }
  #pragma unroll
  for (int n = 0; n < 8; ++n) {
    int col = c0 + n * 16 + l16;
    float bv = bias[col];
    #pragma unroll
    for (int r = 0; r < 8; ++r) {
      int mr = row0 + r + 8 * half;
      float v = acc[n][r] + bv;
      if (OUT_BF16) outb[(size_t)mr * Hsz + col] = f2bf(v);
      else          outf[(size_t)mr * Hsz + col] = v;
    }
  }
}

// ------------------------------------------------------------------
// k,v last-row projections (only k[0,-1], v[0,-1] are needed): fp32 GEMV
// ------------------------------------------------------------------
__global__ void kv_last_kernel(const float* __restrict__ hs,
                               const float* __restrict__ Wk, const float* __restrict__ bk,
                               const float* __restrict__ Wv, const float* __restrict__ bv,
                               float* __restrict__ kl, float* __restrict__ vl) {
  __shared__ float rk[256], rv[256];
  const int j = blockIdx.x;
  const float* x = hs + (size_t)(Ssz - 1) * Hsz;    // batch 0, seq S-1
  float sk = 0.f, sv = 0.f;
  for (int d = threadIdx.x; d < Hsz; d += 256) {
    float xv = x[d];
    sk += xv * Wk[(size_t)j * Hsz + d];
    sv += xv * Wv[(size_t)j * Hsz + d];
  }
  rk[threadIdx.x] = sk; rv[threadIdx.x] = sv;
  __syncthreads();
  for (int s = 128; s > 0; s >>= 1) {
    if (threadIdx.x < s) { rk[threadIdx.x] += rk[threadIdx.x + s];
                           rv[threadIdx.x] += rv[threadIdx.x + s]; }
    __syncthreads();
  }
  if (threadIdx.x == 0) { kl[j] = rk[0] + bk[j]; vl[j] = rv[0] + bv[j]; }
}

// ------------------------------------------------------------------
// Attention pass A: row max + row sum(exp) over M (streaming K via WMMA)
// grid = ROWS/16, block 256. Wave w handles slots chunk + w*16.
// Q tile staged into LDS with the Tensor Data Mover.
// ------------------------------------------------------------------
__global__ __launch_bounds__(256) void attn_pass_a(
    const unsigned short* __restrict__ Qb, const unsigned short* __restrict__ Kb,
    float* __restrict__ rowmax, float* __restrict__ rowinv) {
  __shared__ unsigned short Qs[16 * Hsz];   // 32 KB
  __shared__ float Sc[16 * 128];            // 8 KB chunk scores
  __shared__ float red[256];
  __shared__ float curmax[16];
  __shared__ float rmax[16], rsum[16];
  const int tid = threadIdx.x, wave = tid >> 5, lane = tid & 31;
  const int half = lane >> 4, l16 = lane & 15;
  const int q0 = blockIdx.x * 16;
  if (tid < 32) {   // wave 0 issues the TDM fill and drains TENSORcnt
    tdm_load_1d((unsigned)(size_t)&Qs[0], Qb + (size_t)q0 * Hsz, 16 * Hsz);
    __builtin_amdgcn_s_wait_tensorcnt(0);
  }
  if (tid < 16) { rmax[tid] = -__builtin_inff(); rsum[tid] = 0.f; }
  __syncthreads();
  for (int c = 0; c < Msz; c += 128) {
    const int s0 = c + wave * 16;
    v8f acc = (v8f){0,0,0,0,0,0,0,0};
    const unsigned short* krow = Kb + (size_t)(s0 + l16) * Hsz;
    if (c + 128 < Msz) __builtin_prefetch(krow + 128 * Hsz, 0, 1);
    for (int k0 = 0; k0 < Hsz; k0 += 32) {
      v16bf a = load_frag(&Qs[l16 * Hsz + k0], half);
      v16bf b = load_frag(krow + k0, half);
      acc = wmma_bf16(a, b, acc);
    }
    #pragma unroll
    for (int r = 0; r < 8; ++r)
      Sc[(r + 8 * half) * 128 + wave * 16 + l16] = acc[r] * QSCALE;
    __syncthreads();
    const int g = tid >> 4, t = tid & 15;     // 16 threads per query row
    float lm = -__builtin_inff();
    #pragma unroll
    for (int j = 0; j < 8; ++j) lm = fmaxf(lm, Sc[g * 128 + t * 8 + j]);
    red[g * 16 + t] = lm;
    __syncthreads();
    if (tid < 16) {
      float cm = -__builtin_inff();
      for (int u = 0; u < 16; ++u) cm = fmaxf(cm, red[tid * 16 + u]);
      float old = rmax[tid], nm = fmaxf(old, cm);
      rmax[tid] = nm;
      rsum[tid] *= __expf(old - nm);
      curmax[tid] = nm;
    }
    __syncthreads();
    float ps = 0.f;
    #pragma unroll
    for (int j = 0; j < 8; ++j) ps += __expf(Sc[g * 128 + t * 8 + j] - curmax[g]);
    red[g * 16 + t] = ps;
    __syncthreads();
    if (tid < 16) { float s = 0.f; for (int u = 0; u < 16; ++u) s += red[tid * 16 + u];
                    rsum[tid] += s; }
    __syncthreads();
  }
  if (tid < 16) { rowmax[q0 + tid] = rmax[tid]; rowinv[q0 + tid] = 1.f / rsum[tid]; }
}

// ------------------------------------------------------------------
// Attention pass B: exact probs, P@V via WMMA, importance partials, counts.
// Wave w: score tile for slots chunk+w*16, then PV for d-slice [w*128, +128).
// ------------------------------------------------------------------
__global__ __launch_bounds__(256) void attn_pass_b(
    const unsigned short* __restrict__ Qb, const unsigned short* __restrict__ Kb,
    const unsigned short* __restrict__ Vt,
    const float* __restrict__ rowmax, const float* __restrict__ rowinv,
    unsigned short* __restrict__ memoutb, float* __restrict__ impPart,
    float* __restrict__ counts) {
  __shared__ unsigned short Qs[16 * Hsz];   // 32 KB
  __shared__ unsigned short Ps[16 * 128];   // 4 KB bf16 probs
  __shared__ float rmx[16], rin[16];
  const int tid = threadIdx.x, wave = tid >> 5, lane = tid & 31;
  const int half = lane >> 4, l16 = lane & 15;
  const int q0 = blockIdx.x * 16;
  if (tid < 32) {   // TDM fill of the Q tile
    tdm_load_1d((unsigned)(size_t)&Qs[0], Qb + (size_t)q0 * Hsz, 16 * Hsz);
    __builtin_amdgcn_s_wait_tensorcnt(0);
  }
  if (tid < 16) { rmx[tid] = rowmax[q0 + tid]; rin[tid] = rowinv[q0 + tid]; }
  __syncthreads();
  v8f oacc[8];
  #pragma unroll
  for (int n = 0; n < 8; ++n) oacc[n] = (v8f){0,0,0,0,0,0,0,0};
  float* impRow = impPart + (size_t)blockIdx.x * Msz;
  for (int c = 0; c < Msz; c += 128) {
    const int s0 = c + wave * 16;
    v8f acc = (v8f){0,0,0,0,0,0,0,0};
    const unsigned short* krow = Kb + (size_t)(s0 + l16) * Hsz;
    for (int k0 = 0; k0 < Hsz; k0 += 32) {
      v16bf a = load_frag(&Qs[l16 * Hsz + k0], half);
      v16bf b = load_frag(krow + k0, half);
      acc = wmma_bf16(a, b, acc);
    }
    float psum = 0.f, pcnt = 0.f;
    #pragma unroll
    for (int r = 0; r < 8; ++r) {
      int m = r + 8 * half;
      float p = __expf(acc[r] * QSCALE - rmx[m]) * rin[m];
      psum += p;
      if (p > 0.01f) pcnt += 1.f;
      Ps[m * 128 + wave * 16 + l16] = f2bf(p);
    }
    // combine the two half-wave row groups -> per-slot sums on lanes 0..15
    float osum = psum + __shfl_xor(psum, 16, 32);
    float ocnt = pcnt + __shfl_xor(pcnt, 16, 32);
    if (half == 0) {
      impRow[s0 + l16] = osum;                       // exclusive -> deterministic
      if (ocnt != 0.f) atomicAdd(&counts[s0 + l16], ocnt);  // exact integer adds
    }
    __syncthreads();
    #pragma unroll
    for (int kt = 0; kt < 128; kt += 32) {
      v16bf a = load_frag(&Ps[l16 * 128 + kt], half);
      #pragma unroll
      for (int n = 0; n < 8; ++n) {
        v16bf b = load_frag(Vt + (size_t)(wave * 128 + n * 16 + l16) * Msz + c + kt, half);
        oacc[n] = wmma_bf16(a, b, oacc[n]);
      }
    }
    __syncthreads();   // protect Ps before next chunk overwrite
  }
  #pragma unroll
  for (int n = 0; n < 8; ++n) {
    #pragma unroll
    for (int r = 0; r < 8; ++r) {
      int mr = q0 + r + 8 * half;
      int col = wave * 128 + n * 16 + l16;
      memoutb[(size_t)mr * Hsz + col] = f2bf(oacc[n][r]);
    }
  }
}

// ------------------------------------------------------------------
// Small helpers: zero, importance reduce, age init, reductions
// ------------------------------------------------------------------
__global__ void zerof_kernel(float* p, int n) {
  int i = blockIdx.x * 256 + threadIdx.x; if (i < n) p[i] = 0.f;
}
__global__ void imp_reduce_kernel(const float* __restrict__ part, float* __restrict__ imp) {
  int m = blockIdx.x * 256 + threadIdx.x;
  float s = 0.f;
  for (int w = 0; w < ROWS / 16; ++w) s += part[(size_t)w * Msz + m];
  imp[m] = s;
}
__global__ void age_init_kernel(const float* __restrict__ age, float* __restrict__ nage) {
  int i = blockIdx.x * 256 + threadIdx.x; if (i < Msz) nage[i] = age[i] + 1.f;
}
__global__ void mean_rowmax_kernel(const float* __restrict__ rowmax, float* out) {
  __shared__ float red[256];
  float s = 0.f;
  for (int i = threadIdx.x; i < ROWS; i += 256) s += rowmax[i];
  red[threadIdx.x] = s; __syncthreads();
  for (int st = 128; st > 0; st >>= 1) {
    if (threadIdx.x < st) red[threadIdx.x] += red[threadIdx.x + st];
    __syncthreads();
  }
  if (threadIdx.x == 0) out[0] = red[0] / (float)ROWS;
}
__global__ void usage_kernel(const float* __restrict__ nage, float* out) {
  __shared__ float red[256];
  float c = 0.f;
  for (int i = threadIdx.x; i < Msz; i += 256) c += (nage[i] > 0.f) ? 1.f : 0.f;
  red[threadIdx.x] = c; __syncthreads();
  for (int st = 128; st > 0; st >>= 1) {
    if (threadIdx.x < st) red[threadIdx.x] += red[threadIdx.x + st];
    __syncthreads();
  }
  if (threadIdx.x == 0) out[0] = red[0] / (float)Msz;
}

// ------------------------------------------------------------------
// Exact top-1024 selection (JAX top_k semantics: descending value, ties by
// ascending index). Radix-select threshold then rank equal keys via scan.
// ------------------------------------------------------------------
__device__ __forceinline__ unsigned selkey(float imp, float age) {
  float v = age + 2.f - imp;                 // (age+1) + (1 - importance)
  unsigned u = __float_as_uint(v);
  unsigned ord = (u & 0x80000000u) ? ~u : (u | 0x80000000u);  // ascending order
  return ~ord;                               // smaller key == larger value
}
__global__ __launch_bounds__(1024) void topk_select_kernel(
    const float* __restrict__ imp, const float* __restrict__ age,
    int* __restrict__ updList) {
  __shared__ unsigned hist[256];
  __shared__ unsigned scan[1024];
  __shared__ unsigned sh_prefix, sh_k, sh_eqbase, sh_cnt;
  const int tid = threadIdx.x;
  if (tid == 0) { sh_prefix = 0u; sh_k = 1024u; }
  unsigned maskHigh = 0u;
  for (int shift = 24; shift >= 0; shift -= 8) {
    if (tid < 256) hist[tid] = 0u;
    __syncthreads();
    unsigned prefix = sh_prefix, k = sh_k;
    for (int m = tid; m < Msz; m += 1024) {
      unsigned key = selkey(imp[m], age[m]);
      if ((key & maskHigh) == prefix) atomicAdd(&hist[(key >> shift) & 255u], 1u);
    }
    __syncthreads();
    if (tid == 0) {
      unsigned cum = 0;
      for (int b = 0; b < 256; ++b) {
        unsigned c = hist[b];
        if (cum + c >= k) { sh_prefix = prefix | ((unsigned)b << shift); sh_k = k - cum; break; }
        cum += c;
      }
    }
    maskHigh |= (0xFFu << shift);
    __syncthreads();
  }
  const unsigned T = sh_prefix;              // threshold key
  const unsigned R = sh_k;                   // # ties (key==T) to take, by index order
  if (tid == 0) { sh_cnt = 0u; sh_eqbase = 0u; }
  __syncthreads();
  for (int base = 0; base < Msz; base += 1024) {
    int m = base + tid;
    unsigned key = selkey(imp[m], age[m]);
    unsigned eq = (key == T) ? 1u : 0u;
    bool lt = (key < T);
    unsigned eqbase = sh_eqbase;
    scan[tid] = eq; __syncthreads();
    for (int off = 1; off < 1024; off <<= 1) {
      unsigned add = (tid >= off) ? scan[tid - off] : 0u;
      __syncthreads();
      scan[tid] += add;
      __syncthreads();
    }
    unsigned rank = eqbase + scan[tid] - eq;   // exclusive rank among equals
    if (lt || (eq && rank < R)) {
      unsigned p = atomicAdd(&sh_cnt, 1u);
      updList[p] = m;
    }
    __syncthreads();
    if (tid == 0) sh_eqbase += scan[1023];
    __syncthreads();
  }
}

__global__ void scatter_kernel(const int* __restrict__ updList,
                               const float* __restrict__ kl, const float* __restrict__ vl,
                               float* __restrict__ nK, float* __restrict__ nV,
                               float* __restrict__ nAge) {
  const int idx = updList[blockIdx.x];
  for (int d = threadIdx.x; d < Hsz; d += 256) {
    nK[(size_t)idx * Hsz + d] = kl[d];
    nV[(size_t)idx * Hsz + d] = vl[d];
  }
  if (threadIdx.x == 0) nAge[idx] = 0.f;
}

// ------------------------------------------------------------------
// d_out layout (floats, reference tuple order)
// ------------------------------------------------------------------
static constexpr size_t OFF0 = 0;                                   // output
static constexpr size_t OFF1 = (size_t)ROWS * Hsz;                  // access_counts
static constexpr size_t OFF2 = OFF1 + Msz;                          // max_scores
static constexpr size_t OFF3 = OFF2 + 1;                            // memory_usage
static constexpr size_t OFF4 = OFF3 + 1;                            // new_keys
static constexpr size_t OFF5 = OFF4 + (size_t)Msz * Hsz;            // new_values
static constexpr size_t OFF6 = OFF5 + (size_t)Msz * Hsz;            // new_age

// ws layout (bytes); total ~198 MB
static constexpr size_t WS_HSB  = 0;
static constexpr size_t WS_WQB  = WS_HSB + (size_t)ROWS * Hsz * 2;
static constexpr size_t WS_WOB  = WS_WQB + (size_t)Hsz * Hsz * 2;
static constexpr size_t WS_QB   = WS_WOB + (size_t)Hsz * Hsz * 2;
static constexpr size_t WS_KB   = WS_QB + (size_t)ROWS * Hsz * 2;
static constexpr size_t WS_VT   = WS_KB + (size_t)Msz * Hsz * 2;
static constexpr size_t WS_MOB  = WS_VT + (size_t)Msz * Hsz * 2;
static constexpr size_t WS_RMAX = WS_MOB + (size_t)ROWS * Hsz * 2;
static constexpr size_t WS_RINV = WS_RMAX + (size_t)ROWS * 4;
static constexpr size_t WS_IMPP = WS_RINV + (size_t)ROWS * 4;
static constexpr size_t WS_IMP  = WS_IMPP + (size_t)(ROWS / 16) * Msz * 4;
static constexpr size_t WS_KL   = WS_IMP + (size_t)Msz * 4;
static constexpr size_t WS_VL   = WS_KL + (size_t)Hsz * 4;
static constexpr size_t WS_UPD  = WS_VL + (size_t)Hsz * 4;

extern "C" void kernel_launch(void* const* d_in, const int* in_sizes, int n_in,
                              void* d_out, int out_size, void* d_ws, size_t ws_size,
                              hipStream_t stream) {
  (void)in_sizes; (void)n_in; (void)out_size; (void)ws_size;
  const float* hs  = (const float*)d_in[0];
  const float* Wq  = (const float*)d_in[1];
  const float* bq  = (const float*)d_in[2];
  const float* Wk  = (const float*)d_in[3];
  const float* bk  = (const float*)d_in[4];
  const float* Wv  = (const float*)d_in[5];
  const float* bv  = (const float*)d_in[6];
  const float* Wo  = (const float*)d_in[7];
  const float* bo  = (const float*)d_in[8];
  const float* mk  = (const float*)d_in[9];
  const float* mv  = (const float*)d_in[10];
  const float* mage= (const float*)d_in[11];
  float* out = (float*)d_out;
  char* ws = (char*)d_ws;

  unsigned short* hsb  = (unsigned short*)(ws + WS_HSB);
  unsigned short* Wqb  = (unsigned short*)(ws + WS_WQB);
  unsigned short* Wob  = (unsigned short*)(ws + WS_WOB);
  unsigned short* Qb   = (unsigned short*)(ws + WS_QB);
  unsigned short* Kb   = (unsigned short*)(ws + WS_KB);
  unsigned short* Vt   = (unsigned short*)(ws + WS_VT);
  unsigned short* Mob  = (unsigned short*)(ws + WS_MOB);
  float* rowmax = (float*)(ws + WS_RMAX);
  float* rowinv = (float*)(ws + WS_RINV);
  float* impP   = (float*)(ws + WS_IMPP);
  float* imp    = (float*)(ws + WS_IMP);
  float* klast  = (float*)(ws + WS_KL);
  float* vlast  = (float*)(ws + WS_VL);
  int*   updL   = (int*)(ws + WS_UPD);

  // 1) bf16 staging
  cvt_bf16_kernel<<<2048, 256, 0, stream>>>(hs,  hsb, (size_t)ROWS * Hsz);
  cvt_bf16_kernel<<<2048, 256, 0, stream>>>(Wq,  Wqb, (size_t)Hsz * Hsz);
  cvt_bf16_kernel<<<2048, 256, 0, stream>>>(Wo,  Wob, (size_t)Hsz * Hsz);
  cvt_bf16_kernel<<<2048, 256, 0, stream>>>(mk,  Kb,  (size_t)Msz * Hsz);
  transpose_cvt_kernel<<<dim3(Msz / 32, Hsz / 32), dim3(32, 8), 0, stream>>>(mv, Vt);

  // 2) Q projection (WMMA) + last-row k/v GEMV
  gemm_kernel<1><<<ROWS / 16, 256, 0, stream>>>(hsb, Wqb, bq, Qb, nullptr);
  kv_last_kernel<<<Hsz, 256, 0, stream>>>(hs, Wk, bk, Wv, bv, klast, vlast);

  // 3) attention pass A (row stats), zero counts, pass B (probs+PV)
  attn_pass_a<<<ROWS / 16, 256, 0, stream>>>(Qb, Kb, rowmax, rowinv);
  zerof_kernel<<<(Msz + 255) / 256, 256, 0, stream>>>(out + OFF1, Msz);
  attn_pass_b<<<ROWS / 16, 256, 0, stream>>>(Qb, Kb, Vt, rowmax, rowinv,
                                             Mob, impP, out + OFF1);
  imp_reduce_kernel<<<Msz / 256, 256, 0, stream>>>(impP, imp);

  // 4) output projection (WMMA, f32 out)
  gemm_kernel<0><<<ROWS / 16, 256, 0, stream>>>(Mob, Wob, bo, nullptr, out + OFF0);

  // 5) memory update
  age_init_kernel<<<Msz / 256, 256, 0, stream>>>(mage, out + OFF6);
  (void)hipMemcpyAsync(out + OFF4, mk, (size_t)Msz * Hsz * sizeof(float),
                       hipMemcpyDeviceToDevice, stream);
  (void)hipMemcpyAsync(out + OFF5, mv, (size_t)Msz * Hsz * sizeof(float),
                       hipMemcpyDeviceToDevice, stream);
  topk_select_kernel<<<1, 1024, 0, stream>>>(imp, mage, updL);
  scatter_kernel<<<Ssz, 256, 0, stream>>>(updL, klast, vlast,
                                          out + OFF4, out + OFF5, out + OFF6);

  // 6) scalar outputs
  mean_rowmax_kernel<<<1, 256, 0, stream>>>(rowmax, out + OFF2);
  usage_kernel<<<1, 256, 0, stream>>>(out + OFF6, out + OFF3);
}
